// PINNLoss_34926674051094
// MI455X (gfx1250) — matrix operand order repaired and checked by reading, not compiled, not run on hardware
//
#include <hip/hip_runtime.h>
#include <math.h>

typedef __attribute__((ext_vector_type(16))) _Float16 v16h;
typedef __attribute__((ext_vector_type(8)))  float    v8f;

#define B_SZ 256
#define T_SZ 4096
#define CH   16      // 2*N channels
#define PP_N 144     // params per batch row
#define TPW  16      // 16-timestep tiles per wave

#if defined(__has_builtin)
#if __has_builtin(__builtin_amdgcn_tanhf)
#define FAST_TANH(x) __builtin_amdgcn_tanhf(x)   // v_tanh_f32 (gfx1250 TRANS op)
#endif
#endif
#ifndef FAST_TANH
#define FAST_TANH(x) tanhf(x)
#endif

// ---------------------------------------------------------------------------
__global__ void pinn_zero(float* out) {
    if (threadIdx.x == 0 && blockIdx.x == 0) out[0] = 0.0f;
}

// ---------------------------------------------------------------------------
// supervised loss: mean over B*160 of (concat(params,ic)_pred - target)^2
// ---------------------------------------------------------------------------
__global__ __launch_bounds__(256) void pinn_sup(
    const float* __restrict__ pp, const float* __restrict__ pt,
    const float* __restrict__ icp, const float* __restrict__ ict,
    float* __restrict__ out)
{
    int idx = blockIdx.x * blockDim.x + threadIdx.x;    // 0 .. 40959
    float v = 0.0f;
    if (idx < B_SZ * 160) {
        int b = idx / 160, j = idx % 160;
        float a = (j < PP_N) ? pp[b * PP_N + j] : icp[b * CH + (j - PP_N)];
        float c = (j < PP_N) ? pt[b * PP_N + j] : ict[b * CH + (j - PP_N)];
        float d = a - c;
        v = d * d * (1.0f / 40960.0f);
    }
    #pragma unroll
    for (int off = 16; off >= 1; off >>= 1) v += __shfl_xor(v, off, 32);
    if ((threadIdx.x & 31) == 0) atomicAdd(out, v);
}

// ---------------------------------------------------------------------------
// main fused kernel: MLP + JVP via WMMA, observables, data & physics losses.
// One wave32 loops over TPW=16 tiles (256 timesteps) of one batch row, so
// the W2 f16 packing, per-row params, and the final atomic are amortized.
// Cross-channel exchange uses ds_bpermute (__shfl) instead of LDS+barrier.
// ---------------------------------------------------------------------------
__global__ __launch_bounds__(256) void pinn_main(
    const float* __restrict__ t,
    const float* __restrict__ x_target,
    const float* __restrict__ params_pred,
    const float* __restrict__ W1,
    const float* __restrict__ b1,
    const float* __restrict__ W2,
    const float* __restrict__ b2,
    float* __restrict__ out)
{
    const int lane  = threadIdx.x & 31;
    const int waveG = blockIdx.x * 8 + (threadIdx.x >> 5);   // 0 .. 4095
    const int b     = waveG >> 4;        // 16 waves per batch row
    const int chunk = waveG & 15;        // 256-timestep chunk within the row

    const int half = lane >> 4;          // lane-group half
    const int col  = lane & 15;          // A row M / B,C column N

    // ---- per-wave invariants -------------------------------------------
    const float* pp   = params_pred + b * PP_N;
    const float  bias = b2[col];
    const float  mu_c = pp[8 + (col & 7)];

    float coef[8];
    float g_or_mu;
    if (col < 8) {
        g_or_mu = pp[col];                                   // g_i
        #pragma unroll
        for (int j = 0; j < 8; ++j) coef[j] = pp[16 + col * 8 + j];            // Pi row
    } else {
        g_or_mu = mu_c;                                      // mu_i
        #pragma unroll
        for (int j = 0; j < 8; ++j) coef[j] = pp[16 + 64 + (col - 8) * 8 + j]; // Gamma row
    }

    // B matrices (W2, 64x16 row-major) packed to f16 once per wave
    v16h bB0, bB1;
    #pragma unroll
    for (int i = 0; i < 16; ++i) {
        const int kb = ((i < 8) ? i : i + 8) + half * 8;     // K in 0..31
        bB0[i] = (_Float16)W2[kb * CH + col];
        bB1[i] = (_Float16)W2[(kb + 32) * CH + col];
    }

    const float* xt_wave = x_target + ((size_t)b * CH + col) * T_SZ
                         + chunk * (TPW * 16) + half * 8;
    const float* t_row   = t + b * T_SZ + chunk * (TPW * 16);

    const int srcbase = ((col < 8) ? 8 : 0) + (half << 4);   // bpermute source block

    float acc_data = 0.0f, acc_phys = 0.0f;

    #pragma unroll 1
    for (int it = 0; it < TPW; ++it) {
        const float* xt_base = xt_wave + it * 16;
        __builtin_prefetch(xt_base + 16, 0, 3);              // next tile, global_prefetch_b8

        const float trow = t_row[it * 16 + col];             // time value for A row M=col

        // ---- build A matrices (h and dh) in f16, two K-blocks over H=64 ----
        v16h aL0, aL1, aT0, aT1;
        #pragma unroll
        for (int i = 0; i < 16; ++i) {
            const int k = ((i < 8) ? i : i + 8) + half * 8;  // K in 0..31
            {
                float w  = W1[k];
                float hh = FAST_TANH(trow * w + b1[k]);
                aL0[i] = (_Float16)hh;
                aT0[i] = (_Float16)((1.0f - hh * hh) * w);
            }
            {
                float w  = W1[k + 32];
                float hh = FAST_TANH(trow * w + b1[k + 32]);
                aL1[i] = (_Float16)hh;
                aT1[i] = (_Float16)((1.0f - hh * hh) * w);
            }
        }

        // ---- WMMA: latent = h@W2, dlatent = dh@W2 (f32 accumulate) ----
        v8f cl = {}; v8f ct = {};
        cl = __builtin_amdgcn_wmma_f32_16x16x32_f16(false, aL0, false, bB0, (short)0, cl, false, false);
        cl = __builtin_amdgcn_wmma_f32_16x16x32_f16(false, aL1, false, bB1, (short)0, cl, false, false);
        ct = __builtin_amdgcn_wmma_f32_16x16x32_f16(false, aT0, false, bB0, (short)0, ct, false, false);
        ct = __builtin_amdgcn_wmma_f32_16x16x32_f16(false, aT1, false, bB1, (short)0, ct, false, false);

        // ---- x_target tile: two b128 loads per lane ----
        float xtv[8];
        {
            float4 xa = ((const float4*)xt_base)[0];
            float4 xb = ((const float4*)xt_base)[1];
            xtv[0] = xa.x; xtv[1] = xa.y; xtv[2] = xa.z; xtv[3] = xa.w;
            xtv[4] = xb.x; xtv[5] = xb.y; xtv[6] = xb.z; xtv[7] = xb.w;
        }

        // ---- observables + data loss ----
        float state_r[8], dstate_r[8];
        #pragma unroll
        for (int r = 0; r < 8; ++r) {
            float lat = cl[r] + bias;
            float dl  = ct[r];
            float st, dst;
            if (col >= 8) {                  // relu(q - mu) + JVP gate
                float s   = lat - mu_c;
                bool  pos = s > 0.0f;
                st  = pos ? s  : 0.0f;
                dst = pos ? dl : 0.0f;
            } else {
                st = lat; dst = dl;
            }
            state_r[r]  = st;
            dstate_r[r] = dst;
            float dr = st - xtv[r];
            acc_data += dr * dr;
        }

        // ---- physics residual: cross-channel sums via ds_bpermute ----
        #pragma unroll
        for (int r = 0; r < 8; ++r) {
            float sum = 0.0f;
            #pragma unroll
            for (int j = 0; j < 8; ++j) {
                float v = __shfl(state_r[r], srcbase + j, 32);
                sum += coef[j] * v;
            }
            float rhs = (col < 8) ? (g_or_mu + sum - state_r[r])
                                  : (sum * (g_or_mu - state_r[r]));
            float prr = dstate_r[r] - rhs;
            acc_phys += prr * prr;
        }
    }

    const float INV = 1.0f / 16777216.0f;    // 1 / (B * 2N * T), both means
    float val = (acc_data + acc_phys) * INV;

    #pragma unroll
    for (int off = 16; off >= 1; off >>= 1) val += __shfl_xor(val, off, 32);
    if (lane == 0) atomicAdd(out, val);
}

// ---------------------------------------------------------------------------
extern "C" void kernel_launch(void* const* d_in, const int* in_sizes, int n_in,
                              void* d_out, int out_size, void* d_ws, size_t ws_size,
                              hipStream_t stream) {
    const float* t   = (const float*)d_in[0];
    const float* xt  = (const float*)d_in[1];
    const float* pp  = (const float*)d_in[2];
    const float* ptg = (const float*)d_in[3];
    const float* icp = (const float*)d_in[4];
    const float* ict = (const float*)d_in[5];
    const float* W1  = (const float*)d_in[6];
    const float* b1  = (const float*)d_in[7];
    const float* W2  = (const float*)d_in[8];
    const float* b2  = (const float*)d_in[9];
    float* out = (float*)d_out;

    pinn_zero<<<1, 1, 0, stream>>>(out);
    pinn_sup<<<160, 256, 0, stream>>>(pp, ptg, icp, ict, out);
    // 4096 waves * 16 tiles = 65536 tiles = B*T/16 ; 8 waves per block
    pinn_main<<<512, 256, 0, stream>>>(t, xt, pp, W1, b1, W2, b2, out);
}
